// SAG_37546604102395
// MI455X (gfx1250) — compile-verified, matrix operand order checked by reading, not run on hardware
//
#include <hip/hip_runtime.h>
#include <hip/hip_bf16.h>

// ---------------------------------------------------------------------------
// SAGPool GNN pipeline for MI455X (gfx1250, wave32).
// - All GEMMs (GCN transforms, MLP 64->512->512, final graph MLP) run on
//   v_wmma_f32_16x16x32_bf16 with f32 accumulation; operands pre-converted to
//   bf16 so fragments are contiguous b128 loads matching the ISA 16-bit
//   A/B register layout. The 64MB bf16 activation matrix is L2-resident
//   (192MB L2), so cross-block A re-reads are nearly free.
// - Irregular edge aggregation: per-edge norm precomputed once (shared by all
//   3 GCN layers), b128 row gathers + f32 global atomics (4 per thread).
// - Self-loop term folded into AGG init (saves one HW pass per layer).
// - Top-k pooling is rank-select (pooled sum is permutation invariant);
//   per-graph rows are staged into LDS with global_load_async_to_lds_b128
//   (ASYNCcnt path) before the O(n^2) rank count / weighted pool.
// ---------------------------------------------------------------------------

typedef __bf16 bf16;
typedef bf16  v16bf __attribute__((ext_vector_type(16)));
typedef bf16  v8bf  __attribute__((ext_vector_type(8)));
typedef float v8f   __attribute__((ext_vector_type(8)));

#define N_NODES 65536
#define N_EDGES 1048576
#define N_GRAPHS 64
#define NPG 1024          // nodes per graph
#define KPOOL 512         // top-k per graph

// async 16B global->LDS copy, one per lane (GV mode: VDST = LDS byte offset,
// VADDR = 64-bit global address). Completed by s_wait_asynccnt 0.
__device__ __forceinline__ void async_g2l_b128(const void* gptr, void* lptr) {
    unsigned lds_off = (unsigned)(unsigned long long)lptr;  // addr[31:0] = LDS offset
    asm volatile("global_load_async_to_lds_b128 %0, %1, off"
                 :: "v"(lds_off), "v"(gptr) : "memory");
}
__device__ __forceinline__ void wait_async0() {
    asm volatile("s_wait_asynccnt 0x0" ::: "memory");
}

// ------------------------------ prep kernels -------------------------------

__global__ __launch_bounds__(256) void k_cast_bf16(const float* __restrict__ in,
                                                   bf16* __restrict__ out, int n) {
    int i = blockIdx.x * blockDim.x + threadIdx.x;
    if (i < n) out[i] = (bf16)in[i];
}

// W[K][Nc] f32 -> Wt[Nc][K] bf16 (coalesced read of W)
__global__ __launch_bounds__(256) void k_wt(const float* __restrict__ W,
                                            bf16* __restrict__ Wt, int K, int Nc) {
    int t = blockIdx.x * blockDim.x + threadIdx.x;
    if (t >= K * Nc) return;
    int k = t / Nc, n = t % Nc;
    Wt[(size_t)n * K + k] = (bf16)W[t];
}

__global__ __launch_bounds__(256) void k_zero_f32(float* __restrict__ p, int n) {
    int i = blockIdx.x * blockDim.x + threadIdx.x;
    if (i < n) p[i] = 0.0f;
}

// ------------------------- degree / normalization --------------------------

__global__ __launch_bounds__(256) void k_deg_scatter(const int* __restrict__ dst,
                                                     float* __restrict__ deg, int nE) {
    int e = blockIdx.x * blockDim.x + threadIdx.x;
    if (e < nE) atomicAdd(&deg[dst[e]], 1.0f);
}

__global__ __launch_bounds__(256) void k_dis(float* __restrict__ d, int n) {
    int i = blockIdx.x * blockDim.x + threadIdx.x;
    if (i < n) d[i] = rsqrtf(d[i] + 1.0f);   // +1 self loop
}

// EN[e] = dis[src]*dis[dst], computed once, reused by all 3 GCN layers.
__global__ __launch_bounds__(256) void k_edge_norm(
    const int* __restrict__ src, const int* __restrict__ dst,
    const float* __restrict__ dis, float* __restrict__ EN, int nE) {
    int e = blockIdx.x * blockDim.x + threadIdx.x;
    if (e < nE) EN[e] = dis[src[e]] * dis[dst[e]];
}

// ------------------------------ WMMA GEMM ----------------------------------
// C[M,Nc] = act(A[M,K](bf16) @ Wt^T + bias), Wt stored [Nc][K] bf16.
// Block: 256 threads = 8 waves: 4 along M (16 rows each), 2 along N.
// Wave computes 16 x (BN/2) strip = WT = BN/32 accumulator tiles.
// Fragment layout (16-bit A/B, ISA 7.12.2): lane L holds row (L&15),
// half = L>>4; element e maps to K = e + (e & 8) + 8*half  ->  two
// contiguous 16-byte loads per fragment.
template <int BN>
__global__ __launch_bounds__(256) void gemm_bf16_wmma(
    const bf16* __restrict__ A, const bf16* __restrict__ Wt,
    const float* __restrict__ bias, float* __restrict__ outF,
    bf16* __restrict__ outB, int M, int K, int Nc, int relu) {
    constexpr int WT = BN / 32;
    const int lane = threadIdx.x & 31;
    const int wave = threadIdx.x >> 5;
    const int mw = wave & 3;
    const int nw = wave >> 2;
    const int m0 = blockIdx.x * 64 + mw * 16;
    const int n0 = blockIdx.y * BN + nw * (BN / 2);
    const int r = lane & 15;
    const int half = lane >> 4;

    v8f acc[WT];
#pragma unroll
    for (int t = 0; t < WT; ++t)
#pragma unroll
        for (int g = 0; g < 8; ++g) acc[t][g] = 0.0f;

    const bf16* arow = A + (size_t)(m0 + r) * K;
    for (int k0 = 0; k0 < K; k0 += 32) {
        v16bf a;
        {
            v8bf lo = *(const v8bf*)(arow + k0 + 8 * half);
            v8bf hi = *(const v8bf*)(arow + k0 + 16 + 8 * half);
#pragma unroll
            for (int i = 0; i < 8; ++i) { a[i] = lo[i]; a[i + 8] = hi[i]; }
        }
#pragma unroll
        for (int t = 0; t < WT; ++t) {
            const bf16* brow = Wt + (size_t)(n0 + t * 16 + r) * K;
            v16bf b;
            v8bf lo = *(const v8bf*)(brow + k0 + 8 * half);
            v8bf hi = *(const v8bf*)(brow + k0 + 16 + 8 * half);
#pragma unroll
            for (int i = 0; i < 8; ++i) { b[i] = lo[i]; b[i + 8] = hi[i]; }
            acc[t] = __builtin_amdgcn_wmma_f32_16x16x32_bf16(
                false, a, false, b, (short)0, acc[t], false, false);
        }
    }

    // Epilogue: acc[t][g] = C[m0 + 8*half + g][n0 + 16*t + r]
#pragma unroll
    for (int t = 0; t < WT; ++t) {
        int col = n0 + t * 16 + r;
        float bv = bias ? bias[col] : 0.0f;
#pragma unroll
        for (int g = 0; g < 8; ++g) {
            int row = m0 + 8 * half + g;
            float v = acc[t][g] + bv;
            if (relu) v = fmaxf(v, 0.0f);
            size_t idx = (size_t)row * Nc + col;
            if (outF) outF[idx] = v;
            if (outB) outB[idx] = (bf16)v;
        }
    }
}

// --------------------------- GCN edge aggregation --------------------------

// AGG init with self-loop + bias: AGG[i] = HW[i]*dis^2 + b[f]
__global__ __launch_bounds__(256) void k_agg_init(
    const float* __restrict__ HW, const float* __restrict__ dis,
    const float* __restrict__ b, float* __restrict__ AGG, int total) {
    int i = blockIdx.x * blockDim.x + threadIdx.x;
    if (i >= total) return;
    int node = i >> 6, f = i & 63;
    float dn = dis[node];
    AGG[i] = HW[i] * dn * dn + b[f];
}

// AGG[dst] += HW[src] * EN[e]; thread = (edge, 4-feature chunk): one b128
// gather + 4 f32 atomics.
__global__ __launch_bounds__(256) void k_gcn_scatter(
    const int* __restrict__ src, const int* __restrict__ dst,
    const float* __restrict__ EN, const float* __restrict__ HW,
    float* __restrict__ AGG, int nE) {
    int t = blockIdx.x * blockDim.x + threadIdx.x;
    if (t >= nE * 16) return;
    int e = t >> 4, c = (t & 15) << 2;
    int s = src[e], d = dst[e];
    float en = EN[e];
    const float4 hv = *(const float4*)(HW + (size_t)s * 64 + c);
    float* out = AGG + (size_t)d * 64 + c;
    atomicAdd(out + 0, hv.x * en);
    atomicAdd(out + 1, hv.y * en);
    atomicAdd(out + 2, hv.z * en);
    atomicAdd(out + 3, hv.w * en);
}

// out = relu(AGG) -> bf16
__global__ __launch_bounds__(256) void k_relu_cvt(
    const float* __restrict__ AGG, bf16* __restrict__ out, int total) {
    int i = blockIdx.x * blockDim.x + threadIdx.x;
    if (i < total) out[i] = (bf16)fmaxf(AGG[i], 0.0f);
}

// ------------------------------ SAG pooling --------------------------------

// wave-per-node dual dot: T[n]=h.Wr_rel, R[n]=h.Wr_root (len 512)
__global__ __launch_bounds__(256) void k_score(
    const float* __restrict__ H, const float* __restrict__ wrel,
    const float* __restrict__ wroot, float* __restrict__ T,
    float* __restrict__ R, int n) {
    int wave = threadIdx.x >> 5, lane = threadIdx.x & 31;
    int node = blockIdx.x * 8 + wave;
    if (node >= n) return;
    const float* h = H + (size_t)node * 512;
    float a = 0.0f, b = 0.0f;
    for (int k = lane; k < 512; k += 32) {
        float v = h[k];
        a += v * wrel[k];
        b += v * wroot[k];
    }
    for (int m = 16; m; m >>= 1) {
        a += __shfl_down(a, m, 32);
        b += __shfl_down(b, m, 32);
    }
    if (lane == 0) { T[node] = a; R[node] = b; }
}

__global__ __launch_bounds__(256) void k_scatter_t(
    const int* __restrict__ src, const int* __restrict__ dst,
    const float* __restrict__ T, float* __restrict__ aggs, int nE) {
    int e = blockIdx.x * blockDim.x + threadIdx.x;
    if (e < nE) atomicAdd(&aggs[dst[e]], T[src[e]]);
}

__global__ __launch_bounds__(256) void k_score_fin(
    const float* __restrict__ aggs, const float* __restrict__ R,
    const float* __restrict__ brel, float* __restrict__ score, int n) {
    int i = blockIdx.x * blockDim.x + threadIdx.x;
    if (i < n) score[i] = tanhf(aggs[i] + brel[0] + R[i]);
}

// per-graph rank select (matches lax.top_k lower-index tie break):
// w[i] = score[i] if rank(i) < KPOOL else 0. Scores staged into LDS via
// async global->LDS copies (256 lanes x 16B = 4KB row).
__global__ __launch_bounds__(256) void k_topk(const float* __restrict__ score,
                                              float* __restrict__ w) {
    __shared__ float s[NPG];
    int g = blockIdx.x;
    const float* sg = score + (size_t)g * NPG;
    async_g2l_b128(sg + threadIdx.x * 4, &s[threadIdx.x * 4]);
    wait_async0();
    __syncthreads();
    for (int i = threadIdx.x; i < NPG; i += 256) {
        float si = s[i];
        int rank = 0;
        for (int j = 0; j < NPG; ++j) {
            float sj = s[j];
            rank += (sj > si) || (sj == si && j < i);
        }
        w[(size_t)g * NPG + i] = (rank < KPOOL) ? si : 0.0f;
    }
}

// pooled[g][f] = sum_node w[node] * H[node][f]  (bf16 out for the final GEMM)
__global__ __launch_bounds__(256) void k_pool(const float* __restrict__ w,
                                              const float* __restrict__ H,
                                              bf16* __restrict__ pooled) {
    __shared__ float wl[NPG];
    int g = blockIdx.x;
    async_g2l_b128(w + (size_t)g * NPG + threadIdx.x * 4, &wl[threadIdx.x * 4]);
    wait_async0();
    __syncthreads();
    int f0 = threadIdx.x, f1 = threadIdx.x + 256;
    float a0 = 0.0f, a1 = 0.0f;
    const float* hb = H + (size_t)g * NPG * 512;
    for (int nd = 0; nd < NPG; ++nd) {
        float wv = wl[nd];
        if (wv != 0.0f) {
            a0 += wv * hb[(size_t)nd * 512 + f0];
            a1 += wv * hb[(size_t)nd * 512 + f1];
        }
    }
    pooled[(size_t)g * 512 + f0] = (bf16)a0;
    pooled[(size_t)g * 512 + f1] = (bf16)a1;
}

// ------------------------------- launcher ----------------------------------

extern "C" void kernel_launch(void* const* d_in, const int* in_sizes, int n_in,
                              void* d_out, int out_size, void* d_ws, size_t ws_size,
                              hipStream_t stream) {
    (void)in_sizes; (void)n_in; (void)out_size; (void)ws_size;
    const int N = N_NODES, E = N_EDGES;

    const float* x      = (const float*)d_in[0];
    const int*   ei     = (const int*)d_in[1];
    const int*   src    = ei;
    const int*   dst    = ei + E;
    const float* W1     = (const float*)d_in[3];
    const float* b1     = (const float*)d_in[4];
    const float* W2     = (const float*)d_in[5];
    const float* b2     = (const float*)d_in[6];
    const float* W3     = (const float*)d_in[7];
    const float* b3     = (const float*)d_in[8];
    const float* Wm1    = (const float*)d_in[9];
    const float* bm1    = (const float*)d_in[10];
    const float* Wm2    = (const float*)d_in[11];
    const float* bm2    = (const float*)d_in[12];
    const float* Wr_rel = (const float*)d_in[13];
    const float* br_rel = (const float*)d_in[14];
    const float* Wr_root= (const float*)d_in[15];
    const float* Wq1    = (const float*)d_in[16];
    const float* bq1    = (const float*)d_in[17];
    const float* Wq2    = (const float*)d_in[18];
    const float* bq2    = (const float*)d_in[19];

    char* wsb = (char*)d_ws;
    size_t o = 0;
    auto take = [&](size_t bytes) -> void* {
        void* p = wsb + o;
        o += (bytes + 255) & ~(size_t)255;
        return p;
    };
    bf16*  XB    = (bf16*)take((size_t)N * 128 * 2);
    bf16*  W1T   = (bf16*)take((size_t)128 * 64 * 2);
    bf16*  W2T   = (bf16*)take((size_t)64 * 64 * 2);
    bf16*  W3T   = (bf16*)take((size_t)64 * 64 * 2);
    bf16*  WM1T  = (bf16*)take((size_t)64 * 512 * 2);
    bf16*  WM2T  = (bf16*)take((size_t)512 * 512 * 2);
    bf16*  WQ1T  = (bf16*)take((size_t)512 * 512 * 2);
    bf16*  WQ2T  = (bf16*)take((size_t)512 * 256 * 2);
    float* DIS   = (float*)take((size_t)N * 4);
    float* EN    = (float*)take((size_t)E * 4);
    float* HW    = (float*)take((size_t)N * 64 * 4);
    float* AGG   = (float*)take((size_t)N * 64 * 4);
    bf16*  HA    = (bf16*)take((size_t)N * 64 * 2);
    bf16*  HB    = (bf16*)take((size_t)N * 64 * 2);
    bf16*  HM    = (bf16*)take((size_t)N * 512 * 2);
    float* H512  = (float*)take((size_t)N * 512 * 4);
    float* Tbuf  = (float*)take((size_t)N * 4);
    float* Rbuf  = (float*)take((size_t)N * 4);
    float* AGGS  = (float*)take((size_t)N * 4);
    float* SCORE = (float*)take((size_t)N * 4);
    float* WSEL  = (float*)take((size_t)N * 4);
    bf16*  POOLED= (bf16*)take((size_t)N_GRAPHS * 512 * 2);
    bf16*  MID   = (bf16*)take((size_t)N_GRAPHS * 512 * 2);

    const int TPB = 256;
    auto blks = [](long long n) { return (unsigned)((n + 255) / 256); };

    // ---- prep: bf16 casts + weight transposes ----
    k_cast_bf16<<<blks((long long)N * 128), TPB, 0, stream>>>(x, XB, N * 128);
    k_wt<<<blks(128 * 64), TPB, 0, stream>>>(W1, W1T, 128, 64);
    k_wt<<<blks(64 * 64), TPB, 0, stream>>>(W2, W2T, 64, 64);
    k_wt<<<blks(64 * 64), TPB, 0, stream>>>(W3, W3T, 64, 64);
    k_wt<<<blks(64 * 512), TPB, 0, stream>>>(Wm1, WM1T, 64, 512);
    k_wt<<<blks(512 * 512), TPB, 0, stream>>>(Wm2, WM2T, 512, 512);
    k_wt<<<blks(512 * 512), TPB, 0, stream>>>(Wq1, WQ1T, 512, 512);
    k_wt<<<blks(512 * 256), TPB, 0, stream>>>(Wq2, WQ2T, 512, 256);

    // ---- degree / symmetric normalization + per-edge norm ----
    k_zero_f32<<<blks(N), TPB, 0, stream>>>(DIS, N);
    k_deg_scatter<<<blks(E), TPB, 0, stream>>>(dst, DIS, E);
    k_dis<<<blks(N), TPB, 0, stream>>>(DIS, N);
    k_edge_norm<<<blks(E), TPB, 0, stream>>>(src, dst, DIS, EN, E);

    // ---- GCN layer 1: x(128) -> 64 ----
    gemm_bf16_wmma<64><<<dim3(N / 64, 1), TPB, 0, stream>>>(
        XB, W1T, nullptr, HW, nullptr, N, 128, 64, 0);
    k_agg_init<<<blks((long long)N * 64), TPB, 0, stream>>>(HW, DIS, b1, AGG, N * 64);
    k_gcn_scatter<<<blks((long long)E * 16), TPB, 0, stream>>>(src, dst, EN, HW, AGG, E);
    k_relu_cvt<<<blks((long long)N * 64), TPB, 0, stream>>>(AGG, HA, N * 64);

    // ---- GCN layer 2 ----
    gemm_bf16_wmma<64><<<dim3(N / 64, 1), TPB, 0, stream>>>(
        HA, W2T, nullptr, HW, nullptr, N, 64, 64, 0);
    k_agg_init<<<blks((long long)N * 64), TPB, 0, stream>>>(HW, DIS, b2, AGG, N * 64);
    k_gcn_scatter<<<blks((long long)E * 16), TPB, 0, stream>>>(src, dst, EN, HW, AGG, E);
    k_relu_cvt<<<blks((long long)N * 64), TPB, 0, stream>>>(AGG, HB, N * 64);

    // ---- GCN layer 3 ----
    gemm_bf16_wmma<64><<<dim3(N / 64, 1), TPB, 0, stream>>>(
        HB, W3T, nullptr, HW, nullptr, N, 64, 64, 0);
    k_agg_init<<<blks((long long)N * 64), TPB, 0, stream>>>(HW, DIS, b3, AGG, N * 64);
    k_gcn_scatter<<<blks((long long)E * 16), TPB, 0, stream>>>(src, dst, EN, HW, AGG, E);
    k_relu_cvt<<<blks((long long)N * 64), TPB, 0, stream>>>(AGG, HA, N * 64);

    // ---- node MLP: 64 -> 512 (relu) -> 512 ----
    gemm_bf16_wmma<128><<<dim3(N / 64, 4), TPB, 0, stream>>>(
        HA, WM1T, bm1, nullptr, HM, N, 64, 512, 1);
    gemm_bf16_wmma<128><<<dim3(N / 64, 4), TPB, 0, stream>>>(
        HM, WM2T, bm2, H512, nullptr, N, 512, 512, 0);

    // ---- SAGPooling score: GraphConv(512->1), tanh ----
    k_score<<<N / 8, TPB, 0, stream>>>(H512, Wr_rel, Wr_root, Tbuf, Rbuf, N);
    k_zero_f32<<<blks(N), TPB, 0, stream>>>(AGGS, N);
    k_scatter_t<<<blks(E), TPB, 0, stream>>>(src, dst, Tbuf, AGGS, E);
    k_score_fin<<<blks(N), TPB, 0, stream>>>(AGGS, Rbuf, br_rel, SCORE, N);

    // ---- top-k select + weighted pool ----
    k_topk<<<N_GRAPHS, TPB, 0, stream>>>(SCORE, WSEL);
    k_pool<<<N_GRAPHS, TPB, 0, stream>>>(WSEL, H512, POOLED);

    // ---- final graph MLP: 512 -> 512 (relu) -> 256 ----
    gemm_bf16_wmma<128><<<dim3(1, 4), TPB, 0, stream>>>(
        POOLED, WQ1T, bq1, nullptr, MID, 64, 512, 512, 1);
    gemm_bf16_wmma<128><<<dim3(1, 2), TPB, 0, stream>>>(
        MID, WQ2T, bq2, (float*)d_out, nullptr, 64, 512, 256, 0);
}
